// ClusterModel_27367531610151
// MI455X (gfx1250) — compile-verified, matrix operand order checked by reading, not run on hardware
//
#include <hip/hip_runtime.h>
#include <math.h>

// ---------------------------------------------------------------------------
// ClusterModel forward for MI455X (gfx1250, wave32).
// Dense linears: v_wmma_f32_16x16x32_bf16, operands stored in bf16 (converted
// once at producers), f32 accumulation. Each wave computes a 16x32 output
// tile (one A fragment shared by two B fragments) to halve X re-read traffic.
// ---------------------------------------------------------------------------

typedef __attribute__((ext_vector_type(16))) __bf16 v16bf;
typedef __attribute__((ext_vector_type(8)))  __bf16 v8bf;
typedef __attribute__((ext_vector_type(8)))  float  v8f;

#define NN    10000      // nodes
#define TT    12         // time steps
#define EE    160000     // edges
#define EDIM  288        // edge feature dim
#define ZDIM  544        // edge-MLP dim (288 + 2*128)

#define GF_BIAS 1
#define GF_ACC  2
#define GF_RELU 4
#define GF_OUTB 8        // write bf16 output (instead of f32)

// ---------------- device helpers ----------------

__device__ __forceinline__ __bf16 f2bf(float f) {
  unsigned u = __float_as_uint(f);
  u += 0x7fffu + ((u >> 16) & 1u);          // round-to-nearest-even
  return __builtin_bit_cast(__bf16, (unsigned short)(u >> 16));
}

__device__ __forceinline__ float sigm(float v) { return 1.0f / (1.0f + expf(-v)); }

// 16x32 bf16 fragment, CDNA5 layout: lane L (h=L/16, r=L%16) holds row r,
// K offsets {8h..8h+7} and {16+8h..16+8h+7} -> two 16B loads, no cvt.
__device__ __forceinline__ v16bf load_fragb(const __bf16* base, int h) {
  v8bf lo = *(const v8bf*)(base + 8 * h);
  v8bf hi = *(const v8bf*)(base + 16 + 8 * h);
  return __builtin_shufflevector(lo, hi, 0, 1, 2, 3, 4, 5, 6, 7,
                                 8, 9, 10, 11, 12, 13, 14, 15);
}

__device__ __forceinline__ void atomicMaxF(float* addr, float v) {
  if (v >= 0.0f) atomicMax((int*)addr, __float_as_int(v));
  else           atomicMin((unsigned int*)addr, __float_as_uint(v));
}

// ---------------- WMMA GEMM:  Out[M,N] = X[M,K] @ W[N,K]^T (+bias)(+=)(relu)
// X, W in bf16 (K-major rows). M%16==0, N%32==0, K%32==0 for every call.
// One wave -> 16x32 output tile: one A fragment feeds two WMMAs.

template <int FLAGS>
__global__ __launch_bounds__(128) void k_gemm(
    const __bf16* __restrict__ X, int ldx,
    const __bf16* __restrict__ Wm, int ldw,
    const float* __restrict__ bias,
    float* __restrict__ Out, __bf16* __restrict__ OutB, int ldo,
    int Mtiles, int K)
{
  int lane = threadIdx.x;                       // 0..31
  int tm   = blockIdx.y * 4 + threadIdx.y;      // wave-uniform M tile
  if (tm >= Mtiles) return;
  int tn = blockIdx.x;                          // 32-wide N tile
  int r = lane & 15, h = lane >> 4;
  const __bf16* xrow  = X  + (size_t)(tm * 16 + r) * ldx;
  const __bf16* wrow0 = Wm + (size_t)(tn * 32 + r) * ldw;   // B lane n = W row n
  const __bf16* wrow1 = wrow0 + (size_t)16 * ldw;
  v8f acc0 = {};
  v8f acc1 = {};
#pragma unroll 2
  for (int k0 = 0; k0 < K; k0 += 32) {
    v16bf a  = load_fragb(xrow  + k0, h);
    v16bf b0 = load_fragb(wrow0 + k0, h);
    v16bf b1 = load_fragb(wrow1 + k0, h);
    acc0 = __builtin_amdgcn_wmma_f32_16x16x32_bf16(false, a, false, b0,
                                                   (short)0, acc0, false, false);
    acc1 = __builtin_amdgcn_wmma_f32_16x16x32_bf16(false, a, false, b1,
                                                   (short)0, acc1, false, false);
  }
  int col0 = tn * 32 + r;
  float bv0 = (FLAGS & GF_BIAS) ? bias[col0] : 0.0f;
  float bv1 = (FLAGS & GF_BIAS) ? bias[col0 + 16] : 0.0f;
#pragma unroll
  for (int i = 0; i < 8; ++i) {                 // D row = tm*16 + 8h + i
    size_t idx = (size_t)(tm * 16 + 8 * h + i) * ldo + col0;
    float v0 = acc0[i] + bv0;
    float v1 = acc1[i] + bv1;
    if (FLAGS & GF_ACC)  { v0 += Out[idx]; v1 += Out[idx + 16]; }
    if (FLAGS & GF_RELU) { v0 = fmaxf(v0, 0.0f); v1 = fmaxf(v1, 0.0f); }
    if (FLAGS & GF_OUTB) { OutB[idx] = f2bf(v0); OutB[idx + 16] = f2bf(v1); }
    else                 { Out[idx] = v0; Out[idx + 16] = v1; }
  }
}

// ---------------- elementwise / gather / segment kernels ----------------

struct QTabs { const float* t[9]; };   // [attr,ipid,time] x [win1,win3,win5]

__global__ void k_qwe_ea(float* __restrict__ ea, __bf16* __restrict__ eab,
                         const int* __restrict__ eattr, QTabs qt, int n)
{
  int id = blockIdx.x * 256 + threadIdx.x;
  if (id >= n) return;
  int e = id / EDIM, c = id % EDIM;
  int grp = c / 96, w = (c % 96) / 32, d = c % 32;
  int kwin = (w == 0) ? 1 : (w == 1 ? 3 : 5);
  int pad = (kwin - 1) / 2;
  int idx = eattr[e * 3 + grp] / 2;
  if (idx > 999) idx = 999;
  const float* tab = qt.t[grp * 3 + w] + (size_t)idx * 32;
  float s = 0.0f;
  for (int i = 0; i < kwin; ++i) {
    int d2 = d + i - pad;
    if (d2 >= 0 && d2 < 32) s += tab[d2];
  }
  s /= (float)kwin;
  ea[id] = s;
  eab[id] = f2bf(s);
}

__global__ void k_f2b(__bf16* __restrict__ dst, const float* __restrict__ src, int n)
{
  int id = blockIdx.x * 256 + threadIdx.x;
  if (id < n) dst[id] = f2bf(src[id]);
}

__global__ void k_gather_xe(__bf16* __restrict__ xe, const int* __restrict__ tokens,
                            const float* __restrict__ emb, int n)
{
  int id = blockIdx.x * 256 + threadIdx.x;
  if (id >= n) return;
  int row = id >> 7, c = id & 127;           // row = node*T + t ; c = f*32 + d
  int tok = tokens[row * 4 + (c >> 5)];
  xe[id] = f2bf(emb[(size_t)tok * 32 + (c & 31)]);
}

__global__ void k_gather_rev(unsigned short* __restrict__ xr,
                             const unsigned short* __restrict__ x,
                             const int* __restrict__ lengths, int n)
{
  int id = blockIdx.x * 256 + threadIdx.x;
  if (id >= n) return;
  int nrow = id / (TT * 128), rem = id % (TT * 128);
  int t = rem >> 7, c = rem & 127;
  int st = lengths[nrow] - 1 - t;
  st = st < 0 ? 0 : (st > TT - 1 ? TT - 1 : st);
  xr[id] = x[((size_t)nrow * TT + st) * 128 + c];
}

__global__ void k_lstm_step(__bf16* __restrict__ h, float* __restrict__ c,
                            float* __restrict__ ysum, const float* __restrict__ gates,
                            const int* __restrict__ lengths, int t, int n)
{
  int id = blockIdx.x * 256 + threadIdx.x;
  if (id >= n) return;
  int node = id >> 7, j = id & 127;
  const float* g = gates + (size_t)node * 512;
  float ig = sigm(g[j]);
  float fg = sigm(g[128 + j]);
  float gg = tanhf(g[256 + j]);
  float og = sigm(g[384 + j]);
  float cn = fg * c[id] + ig * gg;
  float hn = og * tanhf(cn);
  if (t < lengths[node]) { c[id] = cn; h[id] = f2bf(hn); ysum[id] += hn; }
}

__global__ void k_divcat(__bf16* __restrict__ y, const float* __restrict__ yf,
                         const float* __restrict__ yb, const int* __restrict__ lengths,
                         int n)
{
  int id = blockIdx.x * 256 + threadIdx.x;
  if (id >= n) return;
  int node = id >> 8, c = id & 255;
  float v = (c < 128) ? yf[node * 128 + c] : yb[node * 128 + (c - 128)];
  y[id] = f2bf(v / (float)lengths[node]);
}

__global__ void k_relu_b2b(unsigned short* __restrict__ dstp,
                           const unsigned short* __restrict__ srcp, int n)
{
  int id = blockIdx.x * 256 + threadIdx.x;
  if (id < n) {
    unsigned short v = srcp[id];
    dstp[id] = (v & 0x8000u) ? (unsigned short)0 : v;   // relu on bf16 bits
  }
}

__global__ void k_relu_f2b(__bf16* __restrict__ dstp, const float* __restrict__ srcp,
                           int n)
{
  int id = blockIdx.x * 256 + threadIdx.x;
  if (id < n) dstp[id] = f2bf(fmaxf(srcp[id], 0.0f));
}

__global__ void k_add(float* __restrict__ o, const float* __restrict__ a,
                      const float* __restrict__ b, int n)
{
  int id = blockIdx.x * 256 + threadIdx.x;
  if (id < n) o[id] = a[id] + b[id];
}

__global__ void k_add_ea(float* __restrict__ ea, __bf16* __restrict__ eab,
                         const float* __restrict__ r2, int n)
{
  int id = blockIdx.x * 256 + threadIdx.x;
  if (id >= n) return;
  float v = ea[id] + r2[id];
  ea[id] = v;
  eab[id] = f2bf(v);
}

__global__ void k_fill(float* __restrict__ p, float v, int n)
{
  int id = blockIdx.x * 256 + threadIdx.x;
  if (id < n) p[id] = v;
}

__global__ void k_fillb(unsigned short* __restrict__ p, int n)
{
  int id = blockIdx.x * 256 + threadIdx.x;
  if (id < n) p[id] = 0;
}

__global__ void k_score(float* __restrict__ score, const float* __restrict__ q,
                        const float* __restrict__ kk, const float* __restrict__ ep,
                        const int* __restrict__ src, const int* __restrict__ dst,
                        int d, float rs)
{
  int e = blockIdx.x * 256 + threadIdx.x;
  if (e >= EE) return;
  const float* qr = q  + (size_t)dst[e] * d;
  const float* kr = kk + (size_t)src[e] * d;
  const float* er = ep + (size_t)e * d;
  float acc = 0.0f;
  for (int j = 0; j < d; ++j) acc += qr[j] * (kr[j] + er[j]);
  score[e] = acc * rs;
}

__global__ void k_segmax(float* __restrict__ m, const float* __restrict__ score,
                         const int* __restrict__ dst)
{
  int e = blockIdx.x * 256 + threadIdx.x;
  if (e < EE) atomicMaxF(&m[dst[e]], score[e]);
}

__global__ void k_expsum(float* __restrict__ score, float* __restrict__ ssum,
                         const float* __restrict__ m, const int* __restrict__ dst)
{
  int e = blockIdx.x * 256 + threadIdx.x;
  if (e >= EE) return;
  float ex = expf(score[e] - m[dst[e]]);
  score[e] = ex;
  atomicAdd(&ssum[dst[e]], ex);
}

__global__ void k_scatter(float* __restrict__ agg, const float* __restrict__ v,
                          const float* __restrict__ ep, const float* __restrict__ ex,
                          const float* __restrict__ ssum, const int* __restrict__ src,
                          const int* __restrict__ dst, int d, int n)
{
  int id = blockIdx.x * 256 + threadIdx.x;
  if (id >= n) return;
  int e = id / d, j = id % d;
  int dn = dst[e], sn = src[e];
  float alpha = ex[e] / (ssum[dn] + 1e-16f);
  atomicAdd(&agg[(size_t)dn * d + j], (v[(size_t)sn * d + j] + ep[id]) * alpha);
}

__global__ void k_bnstats(const float* __restrict__ X, int rows, int cols,
                          float* __restrict__ sum, float* __restrict__ sq)
{
  int row0 = blockIdx.x * 256;
  int rend = row0 + 256; if (rend > rows) rend = rows;
  for (int c = threadIdx.x; c < cols; c += 256) {
    float s = 0.0f, q = 0.0f;
    for (int r = row0; r < rend; ++r) {
      float v = X[(size_t)r * cols + c];
      s += v; q += v * v;
    }
    atomicAdd(&sum[c], s);
    atomicAdd(&sq[c],  q);
  }
}

__global__ void k_bnapply_relu_b(__bf16* __restrict__ Ob, const float* __restrict__ X,
                                 const float* __restrict__ sum, const float* __restrict__ sq,
                                 const float* __restrict__ g, const float* __restrict__ b,
                                 int rows, int cols, int n)
{
  int id = blockIdx.x * 256 + threadIdx.x;
  if (id >= n) return;
  int c = id % cols;
  float inv = 1.0f / (float)rows;
  float mu = sum[c] * inv;
  float var = sq[c] * inv - mu * mu;
  float v = (X[id] - mu) * rsqrtf(var + 1e-5f) * g[c] + b[c];
  Ob[id] = f2bf(fmaxf(v, 0.0f));
}

__global__ void k_gather_z(float* __restrict__ z, const float* __restrict__ ea,
                           const float* __restrict__ h, const int* __restrict__ src,
                           const int* __restrict__ dst, int n)
{
  int id = blockIdx.x * 256 + threadIdx.x;
  if (id >= n) return;
  int e = id / ZDIM, c = id % ZDIM;
  float v;
  if (c < EDIM)            v = ea[(size_t)e * EDIM + c];
  else if (c < EDIM + 128) v = h[(size_t)src[e] * 128 + (c - EDIM)];
  else                     v = h[(size_t)dst[e] * 128 + (c - EDIM - 128)];
  z[id] = v;
}

// ---------------- host helpers ----------------

static inline unsigned grd(size_t n) { return (unsigned)((n + 255) / 256); }

static void gemm(hipStream_t st, const __bf16* X, int ldx, const __bf16* Wm, int ldw,
                 const float* bias, float* Out, __bf16* OutB, int ldo,
                 int M, int Nn, int K, int flags)
{
  int Mtiles = M / 16;
  dim3 grid(Nn / 32, (Mtiles + 3) / 4);
  dim3 block(32, 4);
  switch (flags) {
    case GF_BIAS:
      k_gemm<GF_BIAS><<<grid, block, 0, st>>>(X, ldx, Wm, ldw, bias, Out, OutB, ldo, Mtiles, K);
      break;
    case GF_BIAS | GF_ACC:
      k_gemm<GF_BIAS | GF_ACC><<<grid, block, 0, st>>>(X, ldx, Wm, ldw, bias, Out, OutB, ldo, Mtiles, K);
      break;
    case GF_BIAS | GF_RELU | GF_OUTB:
      k_gemm<GF_BIAS | GF_RELU | GF_OUTB><<<grid, block, 0, st>>>(X, ldx, Wm, ldw, bias, Out, OutB, ldo, Mtiles, K);
      break;
    default: break;
  }
}

struct TW {
  const __bf16 *qw, *kw, *vw, *ew, *sw;
  const float  *qb, *kb, *vb, *eb, *sb;
};

static void run_tconv(hipStream_t st, const TW& p, const __bf16* xin, int din, int dout,
                      const __bf16* eab, const int* src, const int* dst, float* out,
                      float* q_, float* k_, float* v_, float* ep,
                      float* score, float* mseg, float* ssum)
{
  gemm(st, xin, din, p.qw, din, p.qb, q_, nullptr, dout, NN, dout, din, GF_BIAS);
  gemm(st, xin, din, p.kw, din, p.kb, k_, nullptr, dout, NN, dout, din, GF_BIAS);
  gemm(st, xin, din, p.vw, din, p.vb, v_, nullptr, dout, NN, dout, din, GF_BIAS);
  gemm(st, eab, EDIM, p.ew, EDIM, p.eb, ep, nullptr, dout, EE, dout, EDIM, GF_BIAS);
  k_fill<<<grd(NN), 256, 0, st>>>(mseg, -INFINITY, NN);
  k_fill<<<grd(NN), 256, 0, st>>>(ssum, 0.0f, NN);
  k_fill<<<grd((size_t)NN * dout), 256, 0, st>>>(out, 0.0f, NN * dout);
  k_score<<<grd(EE), 256, 0, st>>>(score, q_, k_, ep, src, dst, dout,
                                   1.0f / sqrtf((float)dout));
  k_segmax<<<grd(EE), 256, 0, st>>>(mseg, score, dst);
  k_expsum<<<grd(EE), 256, 0, st>>>(score, ssum, mseg, dst);
  k_scatter<<<grd((size_t)EE * dout), 256, 0, st>>>(out, v_, ep, score, ssum,
                                                    src, dst, dout, EE * dout);
  gemm(st, xin, din, p.sw, din, p.sb, out, nullptr, dout, NN, dout, din, GF_BIAS | GF_ACC);
}

static void run_lstm(hipStream_t st, const __bf16* xseq, const __bf16* wih,
                     const __bf16* whh, const float* bih, const float* bhh,
                     const int* lengths, __bf16* hb, float* cst, float* ysum,
                     float* gates)
{
  k_fillb<<<grd(NN * 128), 256, 0, st>>>((unsigned short*)hb, NN * 128);
  k_fill<<<grd(NN * 128), 256, 0, st>>>(cst, 0.0f, NN * 128);
  k_fill<<<grd(NN * 128), 256, 0, st>>>(ysum, 0.0f, NN * 128);
  for (int t = 0; t < TT; ++t) {
    gemm(st, xseq + t * 128, TT * 128, wih, 128, bih, gates, nullptr, 512,
         NN, 512, 128, GF_BIAS);
    gemm(st, hb, 128, whh, 128, bhh, gates, nullptr, 512,
         NN, 512, 128, GF_BIAS | GF_ACC);
    k_lstm_step<<<grd(NN * 128), 256, 0, st>>>(hb, cst, ysum, gates, lengths, t,
                                               NN * 128);
  }
}

static void run_bn_relu_b(hipStream_t st, __bf16* Ob, const float* Xb,
                          const float* g, const float* b, float* bnsum, float* bnsq)
{
  k_fill<<<grd(ZDIM), 256, 0, st>>>(bnsum, 0.0f, ZDIM);
  k_fill<<<grd(ZDIM), 256, 0, st>>>(bnsq,  0.0f, ZDIM);
  k_bnstats<<<grd(EE), 256, 0, st>>>(Xb, EE, ZDIM, bnsum, bnsq);
  k_bnapply_relu_b<<<grd((size_t)EE * ZDIM), 256, 0, st>>>(Ob, Xb, bnsum, bnsq, g, b,
                                                           EE, ZDIM, EE * ZDIM);
}

// ---------------- entry ----------------
// Param leaves assumed flattened in setup_inputs() recursive insertion order:
//  0 tokens  1 lengths  2 edge_index  3 edge_attr
//  4 emb  5 base.w 6 base.b
//  7..10  lstm_f {wih,whh,bih,bhh}   11..14 lstm_b
//  15..17 qwe_attr[3]  18..20 qwe_ipid[3]  21..23 qwe_time[3]
//  24..33 gcn1.c1 {q,k,v,e,s}x{w,b}  34..43 gcn1.c2  44..45 gcn1.id {w,b}
//  46..47 bn1{g,b} 48..49 lin1{w,b} 50..51 bn2{g,b} 52..53 lin2{w,b}
//  54..63 gcn3.c1  64..73 gcn3.c2

extern "C" void kernel_launch(void* const* d_in, const int* in_sizes, int n_in,
                              void* d_out, int out_size, void* d_ws, size_t ws_size,
                              hipStream_t stream)
{
  (void)in_sizes; (void)n_in; (void)out_size; (void)ws_size;
  const int* tokens  = (const int*)d_in[0];
  const int* lengths = (const int*)d_in[1];
  const int* eindex  = (const int*)d_in[2];
  const int* eattr   = (const int*)d_in[3];
  const int* src = eindex;
  const int* dst = eindex + EE;
  auto F = [&](int i) { return (const float*)d_in[i]; };

  // ---- workspace (byte allocator); peak ~0.94 GB ----
  char* wsp = (char*)d_ws;
  size_t off = 0;
  auto alloc = [&](size_t bytes) {
    char* p = wsp + off;
    off += (bytes + 1023) & ~(size_t)1023;
    return (void*)p;
  };
  float*  ea    = (float*) alloc((size_t)EE * EDIM * 4);   // f32, whole run
  __bf16* eab   = (__bf16*)alloc((size_t)EE * EDIM * 2);   // bf16 mirror
  __bf16* x     = (__bf16*)alloc((size_t)NN * TT * 128 * 2);
  float*  gates = (float*) alloc((size_t)NN * 512 * 4);
  __bf16* hb    = (__bf16*)alloc((size_t)NN * 128 * 2);
  float*  cst   = (float*) alloc((size_t)NN * 128 * 4);
  float*  yf    = (float*) alloc((size_t)NN * 128 * 4);
  float*  yb    = (float*) alloc((size_t)NN * 128 * 4);
  __bf16* y     = (__bf16*)alloc((size_t)NN * 256 * 2);
  __bf16* xact  = (__bf16*)alloc((size_t)NN * 256 * 2);
  __bf16* t1b   = (__bf16*)alloc((size_t)NN * 256 * 2);
  float*  qf    = (float*) alloc((size_t)NN * 256 * 4);
  float*  kf    = (float*) alloc((size_t)NN * 256 * 4);
  float*  vf    = (float*) alloc((size_t)NN * 256 * 4);
  float*  t1    = (float*) alloc((size_t)NN * 256 * 4);
  float*  t2    = (float*) alloc((size_t)NN * 128 * 4);
  float*  hbuf  = (float*) alloc((size_t)NN * 128 * 4);
  float*  score = (float*) alloc((size_t)EE * 4);
  float*  mseg  = (float*) alloc((size_t)NN * 4);
  float*  ssum  = (float*) alloc((size_t)NN * 4);
  float*  bnsum = (float*) alloc((size_t)ZDIM * 4);
  float*  bnsq  = (float*) alloc((size_t)ZDIM * 4);
  // big f32 region: z / r1 / r2 / eproj (time-multiplexed)
  float*  ZF    = (float*) alloc((size_t)EE * ZDIM * 4);
  // big bf16 region: zb / r1b / xe / xr (time-multiplexed)
  __bf16* ZB    = (__bf16*)alloc((size_t)EE * ZDIM * 2);
  float*  zf32  = ZF;           // E x 544 (phase F), also lin1-out, lin2-out
  float*  eproj = ZF;           // E x dout (phases E/G)
  __bf16* zb    = ZB;           // E x 544 bf16 (bn1 out / bn2 out)
  __bf16* xe    = ZB;           // N*T x 128 bf16 (phase B)
  __bf16* xr    = ZB;           // N*T x 128 bf16 (phase C2)

  // ---- bf16 weight copies (one-time per launch) ----
  static const int    widx[28] = {5, 7, 8, 11, 12,
                                  24, 26, 28, 30, 32,
                                  34, 36, 38, 40, 42, 44,
                                  48, 52,
                                  54, 56, 58, 60, 62,
                                  64, 66, 68, 70, 72};
  static const int    wsz[28]  = {16384, 65536, 65536, 65536, 65536,
                                  65536, 65536, 65536, 73728, 65536,
                                  32768, 32768, 32768, 36864, 32768, 32768,
                                  295936, 156672,
                                  16384, 16384, 16384, 36864, 16384,
                                  16384, 16384, 16384, 36864, 16384};
  __bf16* WB[80] = {};
  for (int i = 0; i < 28; ++i) {
    WB[widx[i]] = (__bf16*)alloc((size_t)wsz[i] * 2);
    k_f2b<<<grd(wsz[i]), 256, 0, stream>>>(WB[widx[i]], F(widx[i]), wsz[i]);
  }

  // A) edge features ea = concat(qwe_attr, qwe_ipid, qwe_time)  (+bf16 mirror)
  QTabs qt; for (int i = 0; i < 9; ++i) qt.t[i] = F(15 + i);
  k_qwe_ea<<<grd((size_t)EE * EDIM), 256, 0, stream>>>(ea, eab, eattr, qt, EE * EDIM);

  // B) x = relu(base(emb[tokens]))  -> bf16
  k_gather_xe<<<grd((size_t)NN * TT * 128), 256, 0, stream>>>(xe, tokens, F(4),
                                                              NN * TT * 128);
  gemm(stream, xe, 128, WB[5], 128, F(6), nullptr, x, 128, NN * TT, 128, 128,
       GF_BIAS | GF_RELU | GF_OUTB);

  // C) bidirectional masked LSTM, summed over time
  run_lstm(stream, x, WB[7], WB[8], F(9), F(10), lengths, hb, cst, yf, gates);
  k_gather_rev<<<grd((size_t)NN * TT * 128), 256, 0, stream>>>(
      (unsigned short*)xr, (const unsigned short*)x, lengths, NN * TT * 128);
  run_lstm(stream, xr, WB[11], WB[12], F(13), F(14), lengths, hb, cst, yb, gates);

  // D) y = concat(yf, yb) / lengths  -> bf16
  k_divcat<<<grd((size_t)NN * 256), 256, 0, stream>>>(y, yf, yb, lengths, NN * 256);

  // E) gcn1 resblock: h = id(y) + c2(relu(c1(relu(y))))
  k_relu_b2b<<<grd((size_t)NN * 256), 256, 0, stream>>>(
      (unsigned short*)xact, (const unsigned short*)y, NN * 256);
  TW c11 = {WB[24], WB[26], WB[28], WB[30], WB[32], F(25), F(27), F(29), F(31), F(33)};
  run_tconv(stream, c11, xact, 256, 256, eab, src, dst, t1, qf, kf, vf, eproj,
            score, mseg, ssum);
  k_relu_f2b<<<grd((size_t)NN * 256), 256, 0, stream>>>(t1b, t1, NN * 256);
  TW c12 = {WB[34], WB[36], WB[38], WB[40], WB[42], F(35), F(37), F(39), F(41), F(43)};
  run_tconv(stream, c12, t1b, 256, 128, eab, src, dst, t2, qf, kf, vf, eproj,
            score, mseg, ssum);
  gemm(stream, y, 256, WB[44], 256, F(45), hbuf, nullptr, 128, NN, 128, 256, GF_BIAS);
  k_add<<<grd((size_t)NN * 128), 256, 0, stream>>>(hbuf, hbuf, t2, NN * 128);

  // F) edge MLP: ea2 = ea + lin2(relu(bn2(lin1(relu(bn1([ea,h_src,h_dst]))))))
  k_gather_z<<<grd((size_t)EE * ZDIM), 256, 0, stream>>>(zf32, ea, hbuf, src, dst,
                                                         EE * ZDIM);
  run_bn_relu_b(stream, zb, zf32, F(46), F(47), bnsum, bnsq);        // zb = bn1(z)
  gemm(stream, zb, ZDIM, WB[48], ZDIM, F(49), zf32, nullptr, ZDIM,   // r1 over z
       EE, ZDIM, ZDIM, GF_BIAS);
  run_bn_relu_b(stream, zb, zf32, F(50), F(51), bnsum, bnsq);        // r1b over zb
  gemm(stream, zb, ZDIM, WB[52], ZDIM, F(53), zf32, nullptr, EDIM,   // r2 over r1
       EE, EDIM, ZDIM, GF_BIAS);
  k_add_ea<<<grd((size_t)EE * EDIM), 256, 0, stream>>>(ea, eab, zf32, EE * EDIM);

  // G) gcn3 resblock (no id): out = h + c2(relu(c1(relu(h))))
  k_relu_f2b<<<grd((size_t)NN * 128), 256, 0, stream>>>(xact, hbuf, NN * 128);
  TW c31 = {WB[54], WB[56], WB[58], WB[60], WB[62], F(55), F(57), F(59), F(61), F(63)};
  run_tconv(stream, c31, xact, 128, 128, eab, src, dst, t1, qf, kf, vf, eproj,
            score, mseg, ssum);
  k_relu_f2b<<<grd((size_t)NN * 128), 256, 0, stream>>>(t1b, t1, NN * 128);
  TW c32 = {WB[64], WB[66], WB[68], WB[70], WB[72], F(65), F(67), F(69), F(71), F(73)};
  run_tconv(stream, c32, t1b, 128, 128, eab, src, dst, t2, qf, kf, vf, eproj,
            score, mseg, ssum);
  k_add<<<grd((size_t)NN * 128), 256, 0, stream>>>((float*)d_out, hbuf, t2, NN * 128);
}
